// CustomMatMul_82197084111397
// MI455X (gfx1250) — compile-verified, matrix operand order checked by reading, not run on hardware
//
#include <hip/hip_runtime.h>

// ---------------------------------------------------------------------------
// BFP (block-floating-point) GEMM for MI455X (gfx1250).
//  lhs [16,4096,64] f32, rhs [16,64,4096] f32  ->  out [16,4096,4096] f32
//  BFP: blocks of 32 along K, 8-bit signed mantissa, shared exponent.
//  Quantized values are q * 2^(e-7), |q| <= 128  ==> exactly representable in
//  bf16, so v_wmma_f32_16x16x32_bf16 reproduces the reference f32 math.
//  Roofline: 34.4 GFLOP vs 1.07 GB of output stores -> store-bound (~46 us at
//  23.3 TB/s). Quantized operands (16 MB) are kept L2-resident by issuing the
//  C stream as non-temporal stores; input f32 reads are NT loads.
// ---------------------------------------------------------------------------

typedef __attribute__((ext_vector_type(16))) __bf16 v16bf;
typedef __attribute__((ext_vector_type(8)))  float  v8f;
typedef float f32x4 __attribute__((vector_size(16)));   // native vec for NT load
typedef int   i32x4 __attribute__((vector_size(16)));   // async builtin pointee

typedef union {
    uint4 u[2];   // two 16-byte halves (8 bf16 each)
    v16bf v;
} FragBF;

#define HEADS 16
#define MDIM  4096
#define NDIM  4096
#define KDIM  64
#define BFPB  32      // BFP block size
#define TILE_M 128
#define TILE_N 256
#define LDT    72     // LDS row stride in bf16 (144B, 16B-aligned, conflict-free)

#if defined(__gfx1250__) && __has_builtin(__builtin_amdgcn_global_load_async_to_lds_b128)
#define USE_ASYNC_LDS 1
#else
#define USE_ASYNC_LDS 0
#endif

// Builtin signature (from clang diagnostic): param0 = i32x4 addrspace(1)*
// (global source, non-const); param1 = LDS destination; then imm offset, cpol.
typedef __attribute__((address_space(1))) i32x4 g_i32x4;
typedef __attribute__((address_space(3))) i32x4 l_i32x4;

__device__ __forceinline__ g_i32x4* gbl4(const void* p) {
    return (g_i32x4*)(uintptr_t)p;
}
__device__ __forceinline__ l_i32x4* lds4(void* p) {
    // AS3 pointers are 32-bit; low bits of the generic pointer are the LDS offset.
    return (l_i32x4*)(unsigned)(uintptr_t)p;
}

__device__ __forceinline__ unsigned f32_bits(float x) {
    union { float f; unsigned u; } c; c.f = x; return c.u;
}

// Quantize 32 floats sharing one exponent -> bf16 reconstructed values.
__device__ __forceinline__ void bfp_quant32(const float* __restrict__ v,
                                            __bf16* __restrict__ o) {
    float maxabs = 0.f;
#pragma unroll
    for (int i = 0; i < BFPB; ++i) maxabs = fmaxf(maxabs, fabsf(v[i]));
    if (maxabs > 0.f) {
        // floor(log2(maxabs)) == unbiased exponent field for normal floats
        int e = (int)((f32_bits(maxabs) >> 23) & 0xFF) - 127;
        float step    = __builtin_amdgcn_ldexpf(1.0f, e - 7);   // 2^(e-7)
        float invstep = __builtin_amdgcn_ldexpf(1.0f, 7 - e);
#pragma unroll
        for (int i = 0; i < BFPB; ++i) {
            float q = __builtin_rintf(v[i] * invstep);          // round-half-even
            q = fminf(fmaxf(q, -128.0f), 127.0f);
            o[i] = (__bf16)(q * step);                          // exact in bf16
        }
    } else {
#pragma unroll
        for (int i = 0; i < BFPB; ++i) o[i] = (__bf16)0.0f;
    }
}

// ---------------------------------------------------------------------------
// LHS quantizer: blocks are contiguous 32-float runs. One thread per block.
// ---------------------------------------------------------------------------
__global__ __launch_bounds__(256) void bfp_quant_lhs(const float* __restrict__ X,
                                                     __bf16* __restrict__ Q) {
    const int t = blockIdx.x * 256 + threadIdx.x;     // block id, 131072 total
    const size_t base = (size_t)t * BFPB;
    float v[BFPB];
    const f32x4* src = reinterpret_cast<const f32x4*>(X + base);
#pragma unroll
    for (int i = 0; i < BFPB / 4; ++i) {
        f32x4 f = __builtin_nontemporal_load(&src[i]);   // read-once input
        v[i * 4 + 0] = f[0]; v[i * 4 + 1] = f[1];
        v[i * 4 + 2] = f[2]; v[i * 4 + 3] = f[3];
    }
    __bf16 o[BFPB];
    bfp_quant32(v, o);
    uint4* dst = reinterpret_cast<uint4*>(Q + base);  // regular stores: keep in L2
#pragma unroll
    for (int i = 0; i < 4; ++i)
        dst[i] = reinterpret_cast<const uint4*>(o)[i];
}

// ---------------------------------------------------------------------------
// RHS quantizer: fused transpose. rhs [h][k][n] -> Q [h][n][k] bf16.
// Thread t handles (h, kb, n): 32 gathers of stride 4096 (coalesced over n).
// ---------------------------------------------------------------------------
__global__ __launch_bounds__(256) void bfp_quant_rhs(const float* __restrict__ R,
                                                     __bf16* __restrict__ Q) {
    const int t  = blockIdx.x * 256 + threadIdx.x;    // 16*2*4096 total
    const int n  = t & (NDIM - 1);
    const int kb = (t >> 12) & 1;
    const int h  = t >> 13;
    const float* src = R + ((size_t)h * KDIM + kb * BFPB) * NDIM + n;
    float v[BFPB];
#pragma unroll
    for (int i = 0; i < BFPB; ++i)
        v[i] = __builtin_nontemporal_load(&src[(size_t)i * NDIM]);
    __bf16 o[BFPB];
    bfp_quant32(v, o);
    uint4* dst = reinterpret_cast<uint4*>(Q + ((size_t)(h * NDIM + n)) * KDIM + kb * BFPB);
#pragma unroll
    for (int i = 0; i < 4; ++i)
        dst[i] = reinterpret_cast<const uint4*>(o)[i];
}

// ---------------------------------------------------------------------------
// WMMA GEMM: C[h] = Aq[h] (M x K) * Bq[h]^T (N x K stored row-major).
// Workgroup: 8 waves -> 128x256 C tile. Wave: 64x64 via 4x4 accumulators.
// Tile staging uses gfx1250 async global->LDS copies when available.
// ---------------------------------------------------------------------------
__global__ __launch_bounds__(256) void bfp_gemm(const __bf16* __restrict__ Aq,
                                                const __bf16* __restrict__ Bq,
                                                float* __restrict__ C) {
    __shared__ __bf16 sA[TILE_M * LDT];
    __shared__ __bf16 sB[TILE_N * LDT];

    const int h  = blockIdx.z;
    const int m0 = blockIdx.y * TILE_M;
    const int n0 = blockIdx.x * TILE_N;
    const int tid = threadIdx.x;

    const __bf16* Ah = Aq + (size_t)h * MDIM * KDIM;
    const __bf16* Bh = Bq + (size_t)h * NDIM * KDIM;

#if USE_ASYNC_LDS
    // Direct memory->LDS DMA (no VGPR round trip), tracked with ASYNCcnt.
#pragma unroll
    for (int c = tid; c < TILE_M * 8; c += 256) {
        const int row = c >> 3, col = (c & 7) * 8;
        __builtin_amdgcn_global_load_async_to_lds_b128(
            gbl4(Ah + (size_t)(m0 + row) * KDIM + col),
            lds4(&sA[row * LDT + col]), 0, 0);
    }
#pragma unroll
    for (int c = tid; c < TILE_N * 8; c += 256) {
        const int row = c >> 3, col = (c & 7) * 8;
        __builtin_amdgcn_global_load_async_to_lds_b128(
            gbl4(Bh + (size_t)(n0 + row) * KDIM + col),
            lds4(&sB[row * LDT + col]), 0, 0);
    }
#if __has_builtin(__builtin_amdgcn_s_wait_asynccnt)
    __builtin_amdgcn_s_wait_asynccnt(0);
#else
    asm volatile("s_wait_asynccnt 0" ::: "memory");
#endif
#else
    // Fallback: load to VGPRs, store to LDS.
#pragma unroll
    for (int c = tid; c < TILE_M * 8; c += 256) {
        const int row = c >> 3, col = (c & 7) * 8;
        const uint4 val = *reinterpret_cast<const uint4*>(Ah + (size_t)(m0 + row) * KDIM + col);
        *reinterpret_cast<uint4*>(&sA[row * LDT + col]) = val;
    }
#pragma unroll
    for (int c = tid; c < TILE_N * 8; c += 256) {
        const int row = c >> 3, col = (c & 7) * 8;
        const uint4 val = *reinterpret_cast<const uint4*>(Bh + (size_t)(n0 + row) * KDIM + col);
        *reinterpret_cast<uint4*>(&sB[row * LDT + col]) = val;
    }
#endif
    __syncthreads();

    const int wave = tid >> 5;
    const int lane = tid & 31;
    const int l    = lane & 15;        // row within 16
    const int hi   = lane >> 4;        // half-wave select
    const int wm   = (wave >> 2) * 64; // wave M offset in tile (0/64)
    const int wn   = (wave & 3)  * 64; // wave N offset in tile (0..192)

    v8f acc[4][4] = {};

#pragma unroll
    for (int kk = 0; kk < KDIM; kk += 32) {
        // 16-bit A/B fragment layout: lanes 0-15 hold K 0..7 & 16..23 of row l;
        // lanes 16-31 hold K 8..15 & 24..31 of the same row.
        const int kb = kk + hi * 8;
        FragBF a[4], b[4];
#pragma unroll
        for (int i = 0; i < 4; ++i) {
            const __bf16* p = &sA[(wm + i * 16 + l) * LDT + kb];
            a[i].u[0] = *reinterpret_cast<const uint4*>(p);
            a[i].u[1] = *reinterpret_cast<const uint4*>(p + 16);
        }
#pragma unroll
        for (int j = 0; j < 4; ++j) {
            const __bf16* p = &sB[(wn + j * 16 + l) * LDT + kb];
            b[j].u[0] = *reinterpret_cast<const uint4*>(p);
            b[j].u[1] = *reinterpret_cast<const uint4*>(p + 16);
        }
#pragma unroll
        for (int i = 0; i < 4; ++i)
#pragma unroll
            for (int j = 0; j < 4; ++j)
                acc[i][j] = __builtin_amdgcn_wmma_f32_16x16x32_bf16(
                    false, a[i].v, false, b[j].v,
                    (short)0, acc[i][j], false, false);
    }

    // Epilogue: C/D layout -> VGPR r: lanes 0-15 = row r, lanes 16-31 = row r+8.
    // Non-temporal: the 1.07 GB C stream must not evict L2-resident A/B tiles.
    float* Ch = C + (size_t)h * MDIM * NDIM;
#pragma unroll
    for (int i = 0; i < 4; ++i) {
#pragma unroll
        for (int j = 0; j < 4; ++j) {
            const int mbase = m0 + wm + i * 16 + hi * 8;
            const int nn    = n0 + wn + j * 16 + l;
#pragma unroll
            for (int r = 0; r < 8; ++r)
                __builtin_nontemporal_store(acc[i][j][r],
                    &Ch[(size_t)(mbase + r) * NDIM + nn]);
        }
    }
}

// ---------------------------------------------------------------------------
extern "C" void kernel_launch(void* const* d_in, const int* in_sizes, int n_in,
                              void* d_out, int out_size, void* d_ws, size_t ws_size,
                              hipStream_t stream) {
    const float* lhs = (const float*)d_in[0];   // [16,4096,64]
    const float* rhs = (const float*)d_in[1];   // [16,64,4096]
    float* out = (float*)d_out;                 // [16,4096,4096]

    __bf16* Aq = (__bf16*)d_ws;                          // 8 MB
    __bf16* Bq = Aq + (size_t)HEADS * MDIM * KDIM;       // 8 MB

    const int nblk = HEADS * MDIM * KDIM / BFPB;         // 131072 lhs blocks
    bfp_quant_lhs<<<nblk / 256, 256, 0, stream>>>(lhs, Aq);

    const int nrhs = HEADS * 2 * NDIM;                   // 131072 rhs blocks
    bfp_quant_rhs<<<nrhs / 256, 256, 0, stream>>>(rhs, Bq);

    dim3 grid(NDIM / TILE_N, MDIM / TILE_M, HEADS);
    bfp_gemm<<<grid, 256, 0, stream>>>(Aq, Bq, out);
}
